// Model_39676907883010
// MI455X (gfx1250) — compile-verified, matrix operand order checked by reading, not run on hardware
//
#include <hip/hip_runtime.h>
#include <cstdint>
#include <cstddef>

// Problem constants (reference: B=8, C=32, H=W=48, NCONV=2, SCALE=1.0)
#define BB 8
#define CC 32
#define NN 2304            // H*W
#define NT 144             // NN / 16 tiles
#define WPB 4              // waves per block (wave32)
#define MS2 4              // m-segments for fused column-softmax stats
#define BN (BB * NN)
#define SCALE_F 1.0f

typedef __attribute__((ext_vector_type(16))) _Float16 v16h;
typedef __attribute__((ext_vector_type(8)))  _Float16 v8h;
typedef __attribute__((ext_vector_type(8)))  float    v8f;

// Per-lane K index for 16-bit A/B operands of V_WMMA_*_16X16X32 (ISA 7.12.2):
// lanes 0-15 hold K in {0..7} U {16..23}; lanes 16-31 hold {8..15} U {24..31}.
static __device__ __forceinline__ int kmap(int lane, int j) {
  return ((j >> 3) << 4) + ((lane >> 4) << 3) + (j & 7);
}

// Build a 16x32 fp16 A/B operand from an f32 row (4x float4 loads), converting.
static __device__ __forceinline__ v16h load_row_f32(const float* rowp, int h8) {
  float4 f0 = *(const float4*)(rowp + h8);
  float4 f1 = *(const float4*)(rowp + h8 + 4);
  float4 f2 = *(const float4*)(rowp + h8 + 16);
  float4 f3 = *(const float4*)(rowp + h8 + 20);
  v16h a;
  a[0]  = (_Float16)f0.x; a[1]  = (_Float16)f0.y; a[2]  = (_Float16)f0.z; a[3]  = (_Float16)f0.w;
  a[4]  = (_Float16)f1.x; a[5]  = (_Float16)f1.y; a[6]  = (_Float16)f1.z; a[7]  = (_Float16)f1.w;
  a[8]  = (_Float16)f2.x; a[9]  = (_Float16)f2.y; a[10] = (_Float16)f2.z; a[11] = (_Float16)f2.w;
  a[12] = (_Float16)f3.x; a[13] = (_Float16)f3.y; a[14] = (_Float16)f3.z; a[15] = (_Float16)f3.w;
  return a;
}

// Build a 16x32 fp16 operand from a contiguous fp16 row (2x b128 loads).
static __device__ __forceinline__ v16h load_row_f16(const _Float16* rowp, int h8) {
  v8h lo = *(const v8h*)(rowp + h8);
  v8h hi = *(const v8h*)(rowp + h8 + 16);
  v16h r;
#pragma unroll
  for (int j = 0; j < 8; ++j) { r[j] = lo[j]; r[j + 8] = hi[j]; }
  return r;
}

static __device__ __forceinline__ v8f wmma_f16(v16h a, v16h b, v8f c) {
  return __builtin_amdgcn_wmma_f32_16x16x32_f16(false, a, false, b, (short)0, c,
                                                false, false);
}

// ---------------------------------------------------------------------------
// K1: t = tanh(x); q,k,v projections via fp16 WMMA (K = C = 32 in one step).
// Stores q,v transposed [B][C][N] (packed 16B stores), k row-major [B][N][C].
// ---------------------------------------------------------------------------
__global__ __launch_bounds__(WPB * 32) void k_proj(
    const float* __restrict__ x,
    const float* __restrict__ Wq, const float* __restrict__ bq,
    const float* __restrict__ Wk, const float* __restrict__ bk,
    const float* __restrict__ Wv, const float* __restrict__ bv,
    _Float16* __restrict__ qhT, _Float16* __restrict__ khR,
    _Float16* __restrict__ vhT) {
  const int lane = threadIdx.x & 31;
  const int wid  = blockIdx.x * WPB + (threadIdx.x >> 5);
  const int b = wid / NT, nt = wid % NT;
  const int col = lane & 15;
  const int h8  = (lane >> 4) << 3;
  const int n   = nt * 16 + col;

  // A tile: t[n, c] = tanh(x[b, c, n])  (x channel stride = N -> gathers, small)
  const float* xb = x + (size_t)b * CC * NN + n;
  v16h a;
#pragma unroll
  for (int j = 0; j < 16; ++j)
    a[j] = (_Float16)tanhf(xb[(size_t)kmap(lane, j) * NN]);

#pragma unroll
  for (int ot = 0; ot < 2; ++ot) {
    const int o = ot * 16 + col;  // B column / D column
    v16h wq, wk, wv;
#pragma unroll
    for (int j = 0; j < 16; ++j) {
      const int k = kmap(lane, j);
      wq[j] = (_Float16)Wq[o * CC + k];
      wk[j] = (_Float16)Wk[o * CC + k];
      wv[j] = (_Float16)Wv[o * CC + k];
    }
    v8f z = {};
    v8f aq = wmma_f16(a, wq, z);
    v8f ak = wmma_f16(a, wk, z);
    v8f av = wmma_f16(a, wv, z);
    const float biq = bq[o], bik = bk[o], biv = bv[o];

    v8h pq, pv;
#pragma unroll
    for (int r = 0; r < 8; ++r) {
      pq[r] = (_Float16)(aq[r] + biq);
      pv[r] = (_Float16)(av[r] + biv);
    }
    // D row m = r + h8; transposed store [b][o][n]: 8 consecutive fp16 -> b128
    *(v8h*)(qhT + ((size_t)b * CC + o) * NN + nt * 16 + h8) = pq;
    *(v8h*)(vhT + ((size_t)b * CC + o) * NN + nt * 16 + h8) = pv;
#pragma unroll
    for (int r = 0; r < 8; ++r)
      khR[((size_t)b * NN + nt * 16 + h8 + r) * CC + o] = (_Float16)(ak[r] + bik);
  }
}

// ---------------------------------------------------------------------------
// K2: q_t = adj @ q   ([N,N] f32 x [N,C] fp16 -> [N,C] fp16, K-loop of 72x32)
// Dominant HBM stream (adj read once per iteration).
// ---------------------------------------------------------------------------
__global__ __launch_bounds__(WPB * 32) void k_qt(
    const float* __restrict__ adj, const _Float16* __restrict__ qhT,
    _Float16* __restrict__ qtR) {
  const int lane = threadIdx.x & 31;
  const int wid  = blockIdx.x * WPB + (threadIdx.x >> 5);
  const int b = wid / NT, mt = wid % NT;
  const int col = lane & 15;
  const int h8  = (lane >> 4) << 3;
  const int m   = mt * 16 + col;

  const float*    arow = adj + ((size_t)b * NN + m) * NN;
  const _Float16* q0   = qhT + ((size_t)b * CC + col) * NN;       // c-tile 0
  const _Float16* q1   = qhT + ((size_t)b * CC + 16 + col) * NN;  // c-tile 1
  v8f acc0 = {}, acc1 = {};
  for (int kk = 0; kk < NN; kk += 32) {
    __builtin_prefetch(arow + kk + 128, 0, 0);  // stream-ahead on the long row
    v16h a  = load_row_f32(arow + kk, h8);
    v16h b0 = load_row_f16(q0 + kk, h8);
    v16h b1 = load_row_f16(q1 + kk, h8);
    acc0 = wmma_f16(a, b0, acc0);
    acc1 = wmma_f16(a, b1, acc1);
  }
#pragma unroll
  for (int r = 0; r < 8; ++r) {
    qtR[((size_t)b * NN + mt * 16 + h8 + r) * CC + col]      = (_Float16)acc0[r];
    qtR[((size_t)b * NN + mt * 16 + h8 + r) * CC + 16 + col] = (_Float16)acc1[r];
  }
}

// ---------------------------------------------------------------------------
// K3 (fused): column-softmax stats WITHOUT materializing logits.
// Wave owns one n-tile (16 columns) and 1/MS2 of the m-range; recomputes
// l[m,n] = q_t . k (one WMMA per 16x16 tile, D col = n per lane), keeps
// per-lane online (max, sum-exp) over rows, merges lane<->lane+16 at the end.
// ---------------------------------------------------------------------------
__global__ __launch_bounds__(WPB * 32) void k_stats_f(
    const _Float16* __restrict__ qtR, const _Float16* __restrict__ khR,
    float* __restrict__ pmax, float* __restrict__ psum) {
  const int lane = threadIdx.x & 31;
  const int wid  = blockIdx.x * WPB + (threadIdx.x >> 5);
  const int b    = wid / (NT * MS2);
  const int rem  = wid % (NT * MS2);
  const int nt   = rem / MS2;
  const int seg  = rem % MS2;
  const int col  = lane & 15;
  const int h8   = (lane >> 4) << 3;
  const int n    = nt * 16 + col;

  // fixed B operand: k rows for this n-tile (per-lane column n, K = c)
  const v16h bk = load_row_f16(khR + ((size_t)b * NN + n) * CC, h8);

  float mx = -3.0e38f, s = 0.f;
  const int steps = NT / MS2;  // 36
  for (int t = 0; t < steps; ++t) {
    const int m = (seg * steps + t) * 16 + col;  // A row = lane&15
    v16h a = load_row_f16(qtR + ((size_t)b * NN + m) * CC, h8);
    v8f z = {};
    v8f d = wmma_f16(a, bk, z);   // 8 rows of column n in this lane
#pragma unroll
    for (int r = 0; r < 8; ++r) {
      const float v  = d[r] * SCALE_F;
      const float nm = fmaxf(mx, v);
      s  = s * __expf(mx - nm) + __expf(v - nm);
      mx = nm;
    }
  }
  // lanes L and L+16 hold the same column n with complementary row halves
  const float omx = __shfl_xor(mx, 16, 32);
  const float os  = __shfl_xor(s, 16, 32);
  const float nm  = fmaxf(mx, omx);
  const float st  = s * __expf(mx - nm) + os * __expf(omx - nm);
  if (lane < 16) {
    pmax[(size_t)seg * BN + (size_t)b * NN + n] = nm;
    psum[(size_t)seg * BN + (size_t)b * NN + n] = st;
  }
}

__global__ __launch_bounds__(256) void k_stats2(const float* __restrict__ pmax,
                                                const float* __restrict__ psum,
                                                float* __restrict__ colmax,
                                                float* __restrict__ rcolsum) {
  const int cidx = blockIdx.x * 256 + threadIdx.x;   // [0, BN)
  float mx = -3.0e38f;
#pragma unroll
  for (int s = 0; s < MS2; ++s) mx = fmaxf(mx, pmax[(size_t)s * BN + cidx]);
  float sum = 0.f;
#pragma unroll
  for (int s = 0; s < MS2; ++s)
    sum += psum[(size_t)s * BN + cidx] * __expf(pmax[(size_t)s * BN + cidx] - mx);
  colmax[cidx]  = mx;
  rcolsum[cidx] = 1.f / sum;
}

// ---------------------------------------------------------------------------
// K4 (fused): o = softmax(logits) @ v + residual, recomputing logits.
// Per 32-n chunk: compute l^T tiles (A = k rows n, B = this wave's fixed q_t
// columns), exp in registers; two stacked 16x16 D tiles are element-for-
// element the 32x16 fp16 B operand of the o^T GEMM (Ba[j] = j<8?D0[j]:D1[j-8],
// within-lane), so no LDS/shuffle transpose is needed.
// ---------------------------------------------------------------------------
__global__ __launch_bounds__(WPB * 32) void k_out_f(
    const _Float16* __restrict__ qtR, const _Float16* __restrict__ khR,
    const _Float16* __restrict__ vhT, const float* __restrict__ colmax,
    const float* __restrict__ rcolsum, const float* xin, float* xout) {
  const int lane = threadIdx.x & 31;
  const int wid  = blockIdx.x * WPB + (threadIdx.x >> 5);
  const int b = wid / NT, mt = wid % NT;
  const int col = lane & 15;
  const int h8  = (lane >> 4) << 3;

  // fixed B operand for l^T: this wave's q_t columns (m = mt*16+col, K = c)
  const v16h bq = load_row_f16(qtR + ((size_t)b * NN + mt * 16 + col) * CC, h8);
  const float* cmax = colmax + (size_t)b * NN;
  const float* rsum = rcolsum + (size_t)b * NN;
  const _Float16* v0 = vhT + ((size_t)b * CC + col) * NN;       // v^T c-tile 0
  const _Float16* v1 = vhT + ((size_t)b * CC + 16 + col) * NN;  // v^T c-tile 1
  const _Float16* kb = khR + (size_t)b * NN * CC;

  v8f acc0 = {}, acc1 = {};
  for (int nc = 0; nc < NN; nc += 32) {
    // l^T tiles: rows n = nc(+16) + r + h8, cols m (per-lane col)
    v16h ak0 = load_row_f16(kb + (size_t)(nc + col) * CC, h8);
    v16h ak1 = load_row_f16(kb + (size_t)(nc + 16 + col) * CC, h8);
    v8f z = {};
    v8f d0 = wmma_f16(ak0, bq, z);
    v8f d1 = wmma_f16(ak1, bq, z);
    v16h ba;  // exp'd logits as 32x16 B operand, built within-lane
#pragma unroll
    for (int r = 0; r < 8; ++r) {
      const int n0 = nc + h8 + r;
      const int n1 = nc + 16 + h8 + r;
      ba[r]     = (_Float16)(__expf(d0[r] * SCALE_F - cmax[n0]) * rsum[n0]);
      ba[r + 8] = (_Float16)(__expf(d1[r] * SCALE_F - cmax[n1]) * rsum[n1]);
    }
    v16h av0 = load_row_f16(v0 + nc, h8);
    v16h av1 = load_row_f16(v1 + nc, h8);
    acc0 = wmma_f16(av0, ba, acc0);   // o^T rows c = r+h8, cols m
    acc1 = wmma_f16(av1, ba, acc1);   // o^T rows c = 16+r+h8
  }
  // residual add; lanes 0..15 contiguous in n -> coalesced b32 accesses
#pragma unroll
  for (int r = 0; r < 8; ++r) {
    const size_t i0 = ((size_t)b * CC + h8 + r) * NN + mt * 16 + col;
    const size_t i1 = ((size_t)b * CC + 16 + h8 + r) * NN + mt * 16 + col;
    xout[i0] = xin[i0] + acc0[r];
    xout[i1] = xin[i1] + acc1[r];
  }
}

// ---------------------------------------------------------------------------
extern "C" void kernel_launch(void* const* d_in, const int* in_sizes, int n_in,
                              void* d_out, int out_size, void* d_ws, size_t ws_size,
                              hipStream_t stream) {
  (void)in_sizes; (void)n_in; (void)out_size; (void)ws_size;
  const float* x   = (const float*)d_in[0];
  const float* adj = (const float*)d_in[1];
  const float* Wq  = (const float*)d_in[2];
  const float* bq  = (const float*)d_in[3];
  const float* Wk  = (const float*)d_in[4];
  const float* bk  = (const float*)d_in[5];
  const float* Wv  = (const float*)d_in[6];
  const float* bv  = (const float*)d_in[7];
  float* out = (float*)d_out;

  char* ws = (char*)d_ws;
  const size_t szQ = (size_t)BB * NN * CC * sizeof(_Float16);  // 1.18 MB each
  _Float16* qhT = (_Float16*)ws; ws += szQ;                    // q transposed [B][C][N]
  _Float16* khR = (_Float16*)ws; ws += szQ;                    // k row-major [B][N][C]
  _Float16* vhT = (_Float16*)ws; ws += szQ;                    // v transposed [B][C][N]
  _Float16* qtR = (_Float16*)ws; ws += szQ;                    // q_t row-major [B][N][C]
  float* colmax  = (float*)ws; ws += (size_t)BN * sizeof(float);
  float* rcolsum = (float*)ws; ws += (size_t)BN * sizeof(float);
  float* pmax    = (float*)ws; ws += (size_t)MS2 * BN * sizeof(float);
  float* psum    = (float*)ws; ws += (size_t)MS2 * BN * sizeof(float);
  // total workspace: ~6 MB (no materialized logits)

  const dim3 blk(WPB * 32);
  for (int i = 0; i < 2; ++i) {  // NCONV
    const float* xi = (i == 0) ? x : out;
    k_proj<<<(BB * NT) / WPB, blk, 0, stream>>>(
        xi, Wq + i * CC * CC, bq + i * CC, Wk + i * CC * CC, bk + i * CC,
        Wv + i * CC * CC, bv + i * CC, qhT, khR, vhT);
    k_qt<<<(BB * NT) / WPB, blk, 0, stream>>>(adj, qhT, qtR);
    k_stats_f<<<(BB * NT * MS2) / WPB, blk, 0, stream>>>(qtR, khR, pmax, psum);
    k_stats2<<<BN / 256, 256, 0, stream>>>(pmax, psum, colmax, rcolsum);
    k_out_f<<<(BB * NT) / WPB, blk, 0, stream>>>(qtR, khR, vhT, colmax, rcolsum,
                                                 xi, out);
  }
}
